// Predictor_44341242364289
// MI455X (gfx1250) — compile-verified, matrix operand order checked by reading, not run on hardware
//
#include <hip/hip_runtime.h>

#define NN 50000
#define DD 64
#define EE 800000
#define LL 5

typedef __attribute__((ext_vector_type(16))) __bf16    v16bf;
typedef __attribute__((ext_vector_type(2)))  __bf16    v2bf;
typedef __attribute__((ext_vector_type(8)))  float     v8f;
typedef __attribute__((ext_vector_type(8)))  unsigned  v8u;

// two f32 -> packed bf16 pair (clang lowers fptrunc to v_cvt_pk_bf16_f32 on gfx1250)
static __device__ __forceinline__ unsigned f2bf2(float a, float b) {
  v2bf t;
  t[0] = (__bf16)a;
  t[1] = (__bf16)b;
  return __builtin_bit_cast(unsigned, t);
}

// load a 16-bf16 A/B fragment from packed-pair storage:
// pairs [0..3] at base0, pairs [4..7] at base0+8 (i.e. K offset +16 elements)
static __device__ __forceinline__ v16bf load_frag_pairs(const unsigned* base0) {
  v8u w;
  #pragma unroll
  for (int q = 0; q < 4; ++q) { w[q] = base0[q]; w[4 + q] = base0[8 + q]; }
  return __builtin_bit_cast(v16bf, w);
}

// ---------------- zero the aggregation buffer ----------------
__global__ void __launch_bounds__(256) zero_kernel(float4* __restrict__ p) {
  p[blockIdx.x * 256u + threadIdx.x] = make_float4(0.f, 0.f, 0.f, 0.f);
}

// ---------------- edge scatter: agg[dst] += h[src] ----------------
// 16 threads per edge, each handles 4 contiguous floats (float4 gather + 4 atomics)
__global__ void __launch_bounds__(256) scatter_kernel(
    const float* __restrict__ h, const int* __restrict__ src,
    const int* __restrict__ dst, float* __restrict__ agg)
{
  unsigned g = blockIdx.x * 256u + threadIdx.x;   // E*16 is an exact multiple of 256
  unsigned e = g >> 4;
  unsigned c = (g & 15u) * 4u;
  int s = src[e];
  int d = dst[e];
  const float4 v = *(const float4*)(h + (size_t)s * DD + c);
  float* a = agg + (size_t)d * DD + c;
  __hip_atomic_fetch_add(a + 0, v.x, __ATOMIC_RELAXED, __HIP_MEMORY_SCOPE_AGENT);
  __hip_atomic_fetch_add(a + 1, v.y, __ATOMIC_RELAXED, __HIP_MEMORY_SCOPE_AGENT);
  __hip_atomic_fetch_add(a + 2, v.z, __ATOMIC_RELAXED, __HIP_MEMORY_SCOPE_AGENT);
  __hip_atomic_fetch_add(a + 3, v.w, __ATOMIC_RELAXED, __HIP_MEMORY_SCOPE_AGENT);
}

// ---------------- fused GIN MLP layer: hout = relu((h+agg)@W1^T + b1) @ W2^T + b2 ----
// 8 waves / block, each wave computes a 16-node x 64-col tile with bf16 WMMA.
__global__ void __launch_bounds__(256) gin_layer_kernel(
    const float* __restrict__ hin, const float* __restrict__ agg,
    const float* __restrict__ W1, const float* __restrict__ b1,
    const float* __restrict__ W2, const float* __restrict__ b2,
    float* __restrict__ hout, int nTiles)
{
  __shared__ unsigned w1s[DD * DD / 2];   // bf16 pairs, row-major [j][k]
  __shared__ unsigned w2s[DD * DD / 2];
  __shared__ float    b1s[DD];
  __shared__ float    b2s[DD];
  __shared__ __bf16   zs[8][16 * DD];     // per-wave z staging tile (row-major 16x64)

  const int tid = threadIdx.x;
  {
    const float2* W1v = (const float2*)W1;
    const float2* W2v = (const float2*)W2;
    #pragma unroll
    for (int i = 0; i < (DD * DD / 2) / 256; ++i) {   // 8 pairs per thread per matrix
      int p = i * 256 + tid;
      float2 a = W1v[p]; w1s[p] = f2bf2(a.x, a.y);
      float2 b = W2v[p]; w2s[p] = f2bf2(b.x, b.y);
    }
  }
  if (tid < DD) { b1s[tid] = b1[tid]; b2s[tid] = b2[tid]; }
  __syncthreads();

  const int wave = tid >> 5;
  const int lane = tid & 31;
  const int tile = blockIdx.x * 8 + wave;
  if (tile >= nTiles) return;          // whole-wave uniform; after the only barrier

  const int l15  = lane & 15;
  const int half = lane >> 4;          // A layout: lanes 0-15 K {0..7,16..23}, 16-31 K {8..15,24..31}
  const float* hrow = hin + (size_t)(tile * 16 + l15) * DD;
  const float* arow = agg + (size_t)(tile * 16 + l15) * DD;

  v8f acc[4];
  #pragma unroll
  for (int t = 0; t < 4; ++t) {
    float bj = b1s[t * 16 + l15];      // C/D: N = lane&15 (+16t), bias depends only on N
    #pragma unroll
    for (int r = 0; r < 8; ++r) acc[t][r] = bj;
  }

  // GEMM1: (h+agg) @ W1^T, K = 64 as two K=32 WMMA chunks
  #pragma unroll
  for (int c = 0; c < 2; ++c) {
    const int kb0 = c * 32 + half * 8;   // VGPRs 0-3 hold K kb0..kb0+7
    const int kb1 = kb0 + 16;            // VGPRs 4-7 hold K kb1..kb1+7
    v8u aw;
    #pragma unroll
    for (int q = 0; q < 4; ++q) {
      aw[q]     = f2bf2(hrow[kb0 + 2*q]     + arow[kb0 + 2*q],
                        hrow[kb0 + 2*q + 1] + arow[kb0 + 2*q + 1]);
      aw[4 + q] = f2bf2(hrow[kb1 + 2*q]     + arow[kb1 + 2*q],
                        hrow[kb1 + 2*q + 1] + arow[kb1 + 2*q + 1]);
    }
    v16bf afr = __builtin_bit_cast(v16bf, aw);
    #pragma unroll
    for (int t = 0; t < 4; ++t) {
      v16bf bfr = load_frag_pairs(w1s + (((t * 16 + l15) * DD + kb0) >> 1));
      acc[t] = __builtin_amdgcn_wmma_f32_16x16x32_bf16(
          false, afr, false, bfr, (short)0, acc[t], false, false);
    }
  }

  // ReLU + stage z (bf16) into this wave's LDS tile, row-major 16x64
  __bf16* zt = &zs[wave][0];
  #pragma unroll
  for (int t = 0; t < 4; ++t)
    #pragma unroll
    for (int r = 0; r < 8; ++r) {
      float v = acc[t][r];
      v = v > 0.0f ? v : 0.0f;
      zt[(r + half * 8) * DD + t * 16 + l15] = (__bf16)v;
    }
  asm volatile("s_wait_dscnt 0" ::: "memory");   // LDS turnaround within the wave

  // GEMM2: z @ W2^T + b2
  #pragma unroll
  for (int t = 0; t < 4; ++t) {
    float bj = b2s[t * 16 + l15];
    #pragma unroll
    for (int r = 0; r < 8; ++r) acc[t][r] = bj;
  }
  const unsigned* ztp = (const unsigned*)zt;
  #pragma unroll
  for (int c = 0; c < 2; ++c) {
    const int kb0 = c * 32 + half * 8;
    v16bf afr = load_frag_pairs(ztp + ((l15 * DD + kb0) >> 1));
    #pragma unroll
    for (int t = 0; t < 4; ++t) {
      v16bf bfr = load_frag_pairs(w2s + (((t * 16 + l15) * DD + kb0) >> 1));
      acc[t] = __builtin_amdgcn_wmma_f32_16x16x32_bf16(
          false, afr, false, bfr, (short)0, acc[t], false, false);
    }
  }

  // write f32 result: D element (VGPR r, lane) -> row r + 8*half, col 16t + l15
  float* orow = hout + (size_t)(tile * 16 + half * 8) * DD;
  #pragma unroll
  for (int r = 0; r < 8; ++r)
    #pragma unroll
    for (int t = 0; t < 4; ++t)
      orow[r * DD + t * 16 + l15] = acc[t][r];
}

// ---------------- output heads: out[n][j] = relu(h[n]@W1[j]^T + b1[j]) @ W2[j]^T + b2[j]
__global__ void __launch_bounds__(256) heads_kernel(
    const float* __restrict__ h,
    const float* __restrict__ W1, const float* __restrict__ b1,
    const float* __restrict__ W2, const float* __restrict__ b2,
    float* __restrict__ out, int nPairs)
{
  int p = blockIdx.x * 256 + threadIdx.x;
  if (p >= nPairs) return;
  int n = p >> 1, j = p & 1;
  float x[DD];
  const float* hr = h + (size_t)n * DD;
  #pragma unroll
  for (int k = 0; k < DD; ++k) x[k] = hr[k];
  const float* w1 = W1 + j * DD * DD;
  const float* w2 = W2 + j * DD;
  float accum = b2[j];
  for (int k = 0; k < DD; ++k) {
    float s = b1[j * DD + k];
    const float* wr = w1 + k * DD;
    #pragma unroll
    for (int m = 0; m < DD; ++m) s += x[m] * wr[m];
    s = s > 0.0f ? s : 0.0f;
    accum += s * w2[k];
  }
  out[p] = accum;
}

extern "C" void kernel_launch(void* const* d_in, const int* in_sizes, int n_in,
                              void* d_out, int out_size, void* d_ws, size_t ws_size,
                              hipStream_t stream) {
  const float* X   = (const float*)d_in[0];
  const int*   ei  = (const int*)d_in[1];
  const int*   src = ei;
  const int*   dst = ei + EE;
  const float* gW1 = (const float*)d_in[3];
  const float* gb1 = (const float*)d_in[4];
  const float* gW2 = (const float*)d_in[5];
  const float* gb2 = (const float*)d_in[6];
  const float* oW1 = (const float*)d_in[7];
  const float* ob1 = (const float*)d_in[8];
  const float* oW2 = (const float*)d_in[9];
  const float* ob2 = (const float*)d_in[10];

  float* aggb = (float*)d_ws;                 // N*D f32
  float* hA   = aggb + (size_t)NN * DD;       // N*D f32 ping
  float* hB   = hA   + (size_t)NN * DD;       // N*D f32 pong

  const int nTiles     = NN / 16;                       // 3125 (exact)
  const int zeroBlocks = (NN * DD / 4) / 256;           // 3125 (exact)
  const int scatBlocks = (EE * 16) / 256;               // 50000 (exact)
  const int gemmBlocks = (nTiles + 7) / 8;              // 391
  const int headBlocks = (NN * 2 + 255) / 256;          // 391

  const float* cur = X;
  float* nxt = hA;
  for (int i = 0; i < LL; ++i) {
    zero_kernel<<<zeroBlocks, 256, 0, stream>>>((float4*)aggb);
    scatter_kernel<<<scatBlocks, 256, 0, stream>>>(cur, src, dst, aggb);
    gin_layer_kernel<<<gemmBlocks, 256, 0, stream>>>(
        cur, aggb, gW1 + i * DD * DD, gb1 + i * DD,
        gW2 + i * DD * DD, gb2 + i * DD, nxt, nTiles);
    cur = nxt;
    nxt = (nxt == hA) ? hB : hA;
  }
  heads_kernel<<<headBlocks, 256, 0, stream>>>(cur, oW1, ob1, oW2, ob2,
                                               (float*)d_out, NN * 2);
}
